// GraphNet_30425548325109
// MI455X (gfx1250) — matrix operand order repaired and checked by reading, not compiled, not run on hardware
//
#include <hip/hip_runtime.h>
#include <hip/hip_bf16.h>

// ---------------------------------------------------------------------------
// Interaction-network GNN for MI455X (gfx1250, wave32, WMMA).
//
// Roofline: dominant work = batched tall-skinny GEMMs over the dense
// incidence matrices (20.6 GFLOP, >0.5 GB of fp32 incidence data).
// fp32-WMMA (16x16x4) would be compute-bound; converting incidence weights
// (uniform [0,1), exactly f16-representable to 2^-11) to f16 and using
// v_wmma_f32_16x16x32_f16 with f32 accumulation makes the kernel HBM-bound
// (~30us/pass at 23.3 TB/s). Tiny 22->8->1 / 33->8->8 MLPs are VALU epilogues.
// The f32->f16 conversion/transposition is done once via an LDS-tiled
// transpose so both the read and write streams are full-cacheline coalesced.
// ---------------------------------------------------------------------------

typedef __attribute__((ext_vector_type(16))) _Float16 v16h;
typedef __attribute__((ext_vector_type(8)))  _Float16 v8h;
typedef __attribute__((ext_vector_type(8)))  float    v8f;

#define B_   32
#define N_   1024
#define E_   2048
#define D_   3
#define HD_  8
#define DP_  16   // d = 11 padded to 16 (WMMA N dim)

__device__ __forceinline__ float fast_tanh(float x) {
  x = fminf(fmaxf(x, -15.0f), 15.0f);
  float e = __expf(2.0f * x);
  return (e - 1.0f) / (e + 1.0f);
}
__device__ __forceinline__ float sigmoidf_(float x) {
  return 1.0f / (1.0f + __expf(-x));
}

__device__ __forceinline__ v8f wmma_f16(v16h a, v16h b, v8f c) {
  // D = A(16x32 f16) * B(32x16 f16) + C(16x16 f32)
  return __builtin_amdgcn_wmma_f32_16x16x32_f16(
      /*neg_a=*/false, a, /*neg_b=*/false, b,
      /*c_mod=*/(short)0, c, /*reuse_a=*/false, /*reuse_b=*/false);
}

// A-operand tile loader: row-major source.
// ISA 16-bit A 16x32 layout: lane L -> row (L&15); lanes<16 hold K 0..7,16..23;
// lanes>=16 hold K 8..15,24..31  => two contiguous 16B loads at +kA and +kA+16.
__device__ __forceinline__ v16h load_a_tile(const _Float16* rowk0, int kA) {
  v8h lo = *(const v8h*)(rowk0 + kA);
  v8h hi = *(const v8h*)(rowk0 + kA + 16);
  return __builtin_shufflevector(lo, hi, 0,1,2,3,4,5,6,7,8,9,10,11,12,13,14,15);
}

// Scatter one K-row (16 cols, f32) into packed B-operand storage.
// Packed layout: pack[laneBlock*32 + lane][16 halves]; lane L holds column
// (L&15); lanes<16 hold K 0..7,16..23, lanes>=16 hold K 8..15,24..31.
__device__ __forceinline__ void scatter_row_pack(_Float16* pack, long laneBlockBase,
                                                 int r, const float* w16) {
  _Float16* base = pack + laneBlockBase * 16;
  int laneBase = (r & 8) ? 16 : 0;
  int j = (r & 7) | ((r & 16) >> 1);
#pragma unroll
  for (int c = 0; c < 16; ++c)
    base[(size_t)(laneBase + c) * 16 + j] = (_Float16)w16[c];
}

// ---------------------------------------------------------------------------
// Kernel 1: input net  H = [tanh(X@Wi+bi), X] (padded to 16) -> Hf (f32) + Hpack
// ---------------------------------------------------------------------------
__global__ void k_input(const float* __restrict__ X, const float* __restrict__ Wi,
                        const float* __restrict__ bi,
                        float* __restrict__ Hf, _Float16* __restrict__ Hpack) {
  int t = blockIdx.x * blockDim.x + threadIdx.x;
  if (t >= B_ * N_) return;
  int b = t / N_, n = t % N_;
  const float* x = X + (size_t)t * D_;
  float row[16];
#pragma unroll
  for (int j = 0; j < HD_; ++j) {
    float a = bi[j];
#pragma unroll
    for (int i = 0; i < D_; ++i) a += x[i] * Wi[i * HD_ + j];
    row[j] = fast_tanh(a);
  }
  row[8] = x[0]; row[9] = x[1]; row[10] = x[2];
#pragma unroll
  for (int c = 11; c < 16; ++c) row[c] = 0.0f;

  float* hf = Hf + (size_t)t * DP_;
#pragma unroll
  for (int c = 0; c < 16; ++c) hf[c] = row[c];
  scatter_row_pack(Hpack, ((long)b * (N_ / 32) + (n >> 5)) * 32, n & 31, row);
}

// ---------------------------------------------------------------------------
// Kernel 2: Ri/Ro fp32 -> f16 native [N,E] and transposed [E,N].
// LDS-tiled transpose: 64(n) x 32(e) tiles.
//   read : 32 consecutive fp32 along e  = 128 B / row  (coalesced)
//   write: 64 consecutive f16  along n  = 128 B / row  (coalesced)
// LDS row stride 34 halves = 17 dwords, gcd(17,64)=1 -> conflict-free.
// ---------------------------------------------------------------------------
#define TN 64
#define TE 32

__global__ void __launch_bounds__(256)
k_convert(const float* __restrict__ Ri, const float* __restrict__ Ro,
          _Float16* __restrict__ Ri_h, _Float16* __restrict__ Ro_h,
          _Float16* __restrict__ RiT, _Float16* __restrict__ RoT) {
  int b = blockIdx.z >> 1;
  const float* src = (blockIdx.z & 1) ? Ro : Ri;
  _Float16* dN = (blockIdx.z & 1) ? Ro_h : Ri_h;
  _Float16* dT = (blockIdx.z & 1) ? RoT : RiT;
  int e0 = blockIdx.x * TE;
  int n0 = blockIdx.y * TN;

  __shared__ _Float16 tile[TN][TE + 2];

  int tx = threadIdx.x & 31;   // e within tile
  int ty = threadIdx.x >> 5;   // 0..7
#pragma unroll
  for (int i = 0; i < TN / 8; ++i) {
    int n = n0 + ty + 8 * i;
    float v = src[((size_t)b * N_ + n) * E_ + e0 + tx];
    _Float16 h = (_Float16)v;
    dN[((size_t)b * N_ + n) * E_ + e0 + tx] = h;
    tile[ty + 8 * i][tx] = h;
  }
  __syncthreads();

  int tx2 = threadIdx.x & 63;  // n within tile
  int ty2 = threadIdx.x >> 6;  // 0..3
#pragma unroll
  for (int i = 0; i < TE / 4; ++i) {
    int e = e0 + ty2 + 4 * i;
    dT[((size_t)b * E_ + e) * N_ + n0 + tx2] = tile[tx2][ty2 + 4 * i];
  }
}

// ---------------------------------------------------------------------------
// Kernel 3: edge net. One wave per 16-edge tile.
// bo = RoT@H, bi = RiT@H via WMMA over K=N; VALU MLP epilogue -> e,
// packed e*bo / e*bi for the node GEMM.
// ---------------------------------------------------------------------------
__global__ void __launch_bounds__(32)
k_edge(const _Float16* __restrict__ RoT, const _Float16* __restrict__ RiT,
       const _Float16* __restrict__ Hpack,
       const float* __restrict__ We1, const float* __restrict__ be1,
       const float* __restrict__ We2, const float* __restrict__ be2,
       float* __restrict__ e_out, _Float16* __restrict__ boW,
       _Float16* __restrict__ biW) {
  int lane = threadIdx.x;
  int e0 = blockIdx.x * 16;
  int b = blockIdx.y;
  int mrow = lane & 15;
  int kA = (lane & 16) ? 8 : 0;

  const _Float16* aRo = RoT + ((size_t)b * E_ + e0 + mrow) * N_;
  const _Float16* aRi = RiT + ((size_t)b * E_ + e0 + mrow) * N_;
  const _Float16* bp = Hpack + ((size_t)b * (N_ / 32) * 32 + lane) * 16;

  v8f cBo = {};
  v8f cBi = {};
#pragma unroll 4
  for (int kb = 0; kb < N_ / 32; ++kb) {
    __builtin_prefetch(aRo + (kb + 4) * 32, 0, 1);
    __builtin_prefetch(aRi + (kb + 4) * 32, 0, 1);
    v16h a0 = load_a_tile(aRo + kb * 32, kA);
    v16h a1 = load_a_tile(aRi + kb * 32, kA);
    v16h bb = *(const v16h*)(bp + (size_t)kb * 32 * 16);
    cBo = wmma_f16(a0, bb, cBo);
    cBi = wmma_f16(a1, bb, cBi);
  }

  __shared__ float sBo[256];
  __shared__ float sBi[256];
  int mhi = (lane & 16) ? 8 : 0;
#pragma unroll
  for (int r = 0; r < 8; ++r) {           // C layout: VGPR r -> row r(+8)
    sBo[(r + mhi) * 16 + mrow] = cBo[r];
    sBi[(r + mhi) * 16 + mrow] = cBi[r];
  }
  __syncthreads();

  if (lane < 16) {
    float z[22];
#pragma unroll
    for (int i = 0; i < 11; ++i) {
      z[i] = sBo[lane * 16 + i];
      z[11 + i] = sBi[lane * 16 + i];
    }
    float h[8];
#pragma unroll
    for (int j = 0; j < 8; ++j) {
      float a = be1[j];
#pragma unroll
      for (int i = 0; i < 22; ++i) a += z[i] * We1[i * 8 + j];
      h[j] = fast_tanh(a);
    }
    float v = be2[0];
#pragma unroll
    for (int j = 0; j < 8; ++j) v += h[j] * We2[j];
    float ev = sigmoidf_(v);

    int kglob = e0 + lane;
    e_out[(size_t)b * E_ + kglob] = ev;

    float wbo[16], wbi[16];
#pragma unroll
    for (int c = 0; c < 16; ++c) {
      wbo[c] = ev * sBo[lane * 16 + c];   // pad cols 11..15 are exact zeros
      wbi[c] = ev * sBi[lane * 16 + c];
    }
    long lbb = ((long)b * (E_ / 32) + (kglob >> 5)) * 32;
    scatter_row_pack(boW, lbb, kglob & 31, wbo);
    scatter_row_pack(biW, lbb, kglob & 31, wbi);
  }
}

// ---------------------------------------------------------------------------
// Kernel 4: node net. One wave per 16-node tile.
// mi = Ri@(e.bo), mo = Ro@(e.bi) via WMMA over K=E; 33->8->8 MLP epilogue,
// rebuild H=[H',X] and repack Hpack for next edge pass.
// ---------------------------------------------------------------------------
__global__ void __launch_bounds__(32)
k_node(const _Float16* __restrict__ Ri_h, const _Float16* __restrict__ Ro_h,
       const _Float16* __restrict__ boW, const _Float16* __restrict__ biW,
       const float* __restrict__ X,
       const float* __restrict__ Wn1, const float* __restrict__ bn1,
       const float* __restrict__ Wn2, const float* __restrict__ bn2,
       float* __restrict__ Hf, _Float16* __restrict__ Hpack) {
  int lane = threadIdx.x;
  int n0 = blockIdx.x * 16;
  int b = blockIdx.y;
  int mrow = lane & 15;
  int kA = (lane & 16) ? 8 : 0;

  const _Float16* aRi = Ri_h + ((size_t)b * N_ + n0 + mrow) * E_;
  const _Float16* aRo = Ro_h + ((size_t)b * N_ + n0 + mrow) * E_;
  const _Float16* bpo = boW + ((size_t)b * (E_ / 32) * 32 + lane) * 16;
  const _Float16* bpi = biW + ((size_t)b * (E_ / 32) * 32 + lane) * 16;

  v8f cMi = {};
  v8f cMo = {};
#pragma unroll 4
  for (int kb = 0; kb < E_ / 32; ++kb) {
    __builtin_prefetch(aRi + (kb + 4) * 32, 0, 1);
    __builtin_prefetch(aRo + (kb + 4) * 32, 0, 1);
    v16h a0 = load_a_tile(aRi + kb * 32, kA);
    v16h a1 = load_a_tile(aRo + kb * 32, kA);
    v16h b0 = *(const v16h*)(bpo + (size_t)kb * 32 * 16);
    v16h b1 = *(const v16h*)(bpi + (size_t)kb * 32 * 16);
    cMi = wmma_f16(a0, b0, cMi);
    cMo = wmma_f16(a1, b1, cMo);
  }

  __shared__ float sMi[256];
  __shared__ float sMo[256];
  int mhi = (lane & 16) ? 8 : 0;
#pragma unroll
  for (int r = 0; r < 8; ++r) {
    sMi[(r + mhi) * 16 + mrow] = cMi[r];
    sMo[(r + mhi) * 16 + mrow] = cMo[r];
  }
  __syncthreads();

  if (lane < 16) {
    int n = n0 + lane;
    const float* hrow = Hf + ((size_t)b * N_ + n) * DP_;
    float z[33];
#pragma unroll
    for (int i = 0; i < 11; ++i) {
      z[i] = sMi[lane * 16 + i];
      z[11 + i] = sMo[lane * 16 + i];
      z[22 + i] = hrow[i];
    }
    float h[8];
#pragma unroll
    for (int j = 0; j < 8; ++j) {
      float a = bn1[j];
#pragma unroll
      for (int i = 0; i < 33; ++i) a += z[i] * Wn1[i * 8 + j];
      h[j] = fast_tanh(a);
    }
    const float* x = X + ((size_t)b * N_ + n) * D_;
    float row[16];
#pragma unroll
    for (int j = 0; j < 8; ++j) {
      float a = bn2[j];
#pragma unroll
      for (int i = 0; i < 8; ++i) a += h[i] * Wn2[i * 8 + j];
      row[j] = fast_tanh(a);
    }
    row[8] = x[0]; row[9] = x[1]; row[10] = x[2];
#pragma unroll
    for (int c = 11; c < 16; ++c) row[c] = 0.0f;

    float* hf = Hf + ((size_t)b * N_ + n) * DP_;
#pragma unroll
    for (int c = 0; c < 16; ++c) hf[c] = row[c];
    scatter_row_pack(Hpack, ((long)b * (N_ / 32) + (n >> 5)) * 32, n & 31, row);
  }
}

// ---------------------------------------------------------------------------
extern "C" void kernel_launch(void* const* d_in, const int* in_sizes, int n_in,
                              void* d_out, int out_size, void* d_ws, size_t ws_size,
                              hipStream_t stream) {
  (void)in_sizes; (void)n_in; (void)out_size; (void)ws_size;
  const float* X   = (const float*)d_in[0];
  const float* Ri  = (const float*)d_in[1];
  const float* Ro  = (const float*)d_in[2];
  const float* Wi  = (const float*)d_in[3];
  const float* bi  = (const float*)d_in[4];
  const float* We1 = (const float*)d_in[5];
  const float* be1 = (const float*)d_in[6];
  const float* We2 = (const float*)d_in[7];
  const float* be2 = (const float*)d_in[8];
  const float* Wn1 = (const float*)d_in[9];
  const float* bn1 = (const float*)d_in[10];
  const float* Wn2 = (const float*)d_in[11];
  const float* bn2 = (const float*)d_in[12];

  char* ws = (char*)d_ws;
  const size_t SB = (size_t)B_ * E_ * N_ * sizeof(_Float16);  // 128 MB
  _Float16* RiT  = (_Float16*)(ws + 0 * SB);                  // [B][E][N]
  _Float16* RoT  = (_Float16*)(ws + 1 * SB);
  _Float16* Ri_h = (_Float16*)(ws + 2 * SB);                  // [B][N][E]
  _Float16* Ro_h = (_Float16*)(ws + 3 * SB);
  char* p = ws + 4 * SB;
  float*    Hf    = (float*)p;             p += (size_t)B_ * N_ * DP_ * 4;
  _Float16* Hpack = (_Float16*)p;          p += (size_t)B_ * N_ * DP_ * 2;
  _Float16* boW   = (_Float16*)p;          p += (size_t)B_ * E_ * DP_ * 2;
  _Float16* biW   = (_Float16*)p;          p += (size_t)B_ * E_ * DP_ * 2;
  float*    eBuf  = (float*)p;

  k_input<<<(B_ * N_ + 255) / 256, 256, 0, stream>>>(X, Wi, bi, Hf, Hpack);

  dim3 gc(E_ / TE, N_ / TN, 2 * B_);
  k_convert<<<gc, 256, 0, stream>>>(Ri, Ro, Ri_h, Ro_h, RiT, RoT);

  dim3 ge(E_ / 16, B_);
  dim3 gn(N_ / 16, B_);
  for (int it = 0; it < 3; ++it) {
    k_edge<<<ge, 32, 0, stream>>>(RoT, RiT, Hpack, We1, be1, We2, be2,
                                  eBuf, boW, biW);
    k_node<<<gn, 32, 0, stream>>>(Ri_h, Ro_h, boW, biW, X, Wn1, bn1, Wn2, bn2,
                                  Hf, Hpack);
  }
  // Final edge net writes straight to d_out ([B,E] f32).
  k_edge<<<ge, 32, 0, stream>>>(RoT, RiT, Hpack, We1, be1, We2, be2,
                                (float*)d_out, boW, biW);
}
